// VQEncoder_77833397338785
// MI455X (gfx1250) — compile-verified, matrix operand order checked by reading, not run on hardware
//
#include <hip/hip_runtime.h>

// ---------------- types ----------------
typedef __attribute__((ext_vector_type(16))) __bf16 v16bf;
typedef __attribute__((ext_vector_type(8)))  float  v8f;

#define B_    8
#define CIN   1024
#define T_    4096
#define T2    2048
#define DVQ   1024
#define KCB   2048
#define KDIM  2048   // CIN * DS

// LDS row strides (in ushorts): 32B-aligned, not bank-aligned
#define ASTRIDE_K  2064   // KDIM + 16  (4128 B = 32*129)
#define ASTRIDE_D  1040   // DVQ  + 16  (2080 B = 32*65)

__device__ __forceinline__ unsigned short f2bf(float f) {
    unsigned int u = __float_as_uint(f);
    unsigned int r = u + 0x7fffu + ((u >> 16) & 1u);
    return (unsigned short)(r >> 16);
}
__device__ __forceinline__ float bf2f(unsigned short h) {
    return __uint_as_float(((unsigned int)h) << 16);
}

// ---- gfx1250 async copy: global -> LDS, tracked by ASYNCcnt ----
__device__ __forceinline__ void async_b128(unsigned lds_off, const void* gptr) {
    unsigned long long ga = (unsigned long long)(uintptr_t)gptr;
    asm volatile("global_load_async_to_lds_b128 %0, %1, off"
                 :: "v"(lds_off), "v"(ga) : "memory");
}
__device__ __forceinline__ void async_wait0() {
    asm volatile("s_wait_asynccnt 0x0" ::: "memory");
}

__device__ __forceinline__ v8f wmma_bf16(v16bf a, v16bf b, v8f c) {
    return __builtin_amdgcn_wmma_f32_16x16x32_bf16(false, a, false, b, (short)0, c, false, false);
}

// ---------------- K0: pack x (B,C,T) -> Xbf (B,T2,KDIM) bf16, k=c*2+s ----------
__global__ void k_pack_x(const float* __restrict__ x, unsigned short* __restrict__ Xbf) {
    __shared__ float tile[32][65];
    const int b  = blockIdx.z;
    const int c0 = blockIdx.y * 32;
    const int t0 = blockIdx.x * 32;
    const int tid  = threadIdx.x;
    const int row  = tid >> 3;
    const int col8 = tid & 7;
    const float* xp = x + ((size_t)(b * CIN + c0 + row)) * T_ + (size_t)t0 * 2 + col8 * 8;
#pragma unroll
    for (int i = 0; i < 8; ++i) tile[row][col8 * 8 + i] = xp[i];
    __syncthreads();
    unsigned short* op = Xbf + ((size_t)(b * T2) + t0 + row) * KDIM + (size_t)(c0 * 2) + col8 * 8;
#pragma unroll
    for (int i = 0; i < 8; ++i) {
        const int kk = col8 * 8 + i;
        const int cc = kk >> 1, s = kk & 1;
        op[i] = f2bf(tile[cc][row * 2 + s]);
    }
}

// ---------------- K1: fp32 -> bf16 weight conversion ----------------
__global__ void k_cvt_w(const float* __restrict__ win, const float* __restrict__ cb,
                        const float* __restrict__ wout,
                        unsigned short* __restrict__ Wbf, unsigned short* __restrict__ Cbf,
                        unsigned short* __restrict__ Obf) {
    const size_t i = (size_t)blockIdx.x * blockDim.x + threadIdx.x;
    const size_t n1 = (size_t)DVQ * KDIM;
    const size_t n2 = (size_t)KCB * DVQ;
    const size_t n3 = (size_t)CIN * DVQ;
    if (i < n1)                Wbf[i]           = f2bf(win[i]);
    else if (i < n1 + n2)      Cbf[i - n1]      = f2bf(cb[i - n1]);
    else if (i < n1 + n2 + n3) Obf[i - n1 - n2] = f2bf(wout[i - n1 - n2]);
}

// ---------------- K2: cnorm[k] = 0.5*||c_k||^2 ----------------
__global__ void k_cnorm(const float* __restrict__ cb, float* __restrict__ cnorm) {
    __shared__ float red[256];
    const int k = blockIdx.x;
    const float* row = cb + (size_t)k * DVQ;
    float s = 0.f;
    for (int i = threadIdx.x; i < DVQ; i += 256) { float v = row[i]; s += v * v; }
    red[threadIdx.x] = s;
    __syncthreads();
    for (int off = 128; off > 0; off >>= 1) {
        if (threadIdx.x < off) red[threadIdx.x] += red[threadIdx.x + off];
        __syncthreads();
    }
    if (threadIdx.x == 0) cnorm[k] = 0.5f * red[0];
}

// ---------- K3: GEMM1 h = X2 @ W^T + bias -> Hbf; async-staged A, 2x2 tiles ----
// block: 256 thr (8 waves). Covers 32 rows (B*T2) x 256 cols (DVQ).
__global__ void __launch_bounds__(256)
k_gemm_convin(const unsigned short* __restrict__ Xbf, const unsigned short* __restrict__ Wbf,
              const float* __restrict__ bias, unsigned short* __restrict__ Hbf) {
    extern __shared__ unsigned short dynsmem[];
    const int tid  = threadIdx.x;
    const int wave = tid >> 5, lane = tid & 31;
    const int m    = lane & 15, half = lane >> 4;
    const size_t row0 = (size_t)blockIdx.y * 32;
    const int    c0   = blockIdx.x * 256 + wave * 32;

    { // async stage A panel: 32 rows x 4096 B  ->  rows padded to 4128 B
        const unsigned lbase = (unsigned)(uintptr_t)dynsmem;
#pragma unroll 4
        for (int i = 0; i < 32; ++i) {
            const int c   = i * 256 + tid;          // 8192 x 16B chunks
            const int r   = c >> 8;
            const int off = (c & 255) * 16;
            async_b128(lbase + (unsigned)(r * (ASTRIDE_K * 2) + off),
                       (const char*)(Xbf + (row0 + r) * KDIM) + off);
        }
        async_wait0();
    }
    __syncthreads();

    const unsigned short* A0 = dynsmem + (size_t)(0  + m) * ASTRIDE_K + half * 16;
    const unsigned short* A1 = dynsmem + (size_t)(16 + m) * ASTRIDE_K + half * 16;
    const unsigned short* B0 = Wbf + ((size_t)(c0 + m))      * KDIM + half * 16;
    const unsigned short* B1 = Wbf + ((size_t)(c0 + 16 + m)) * KDIM + half * 16;

    v8f acc00 = {0.f,0.f,0.f,0.f,0.f,0.f,0.f,0.f}, acc01 = acc00, acc10 = acc00, acc11 = acc00;
#pragma unroll 2
    for (int kc = 0; kc < KDIM / 32; ++kc) {
        v16bf a0 = *reinterpret_cast<const v16bf*>(A0 + kc * 32);
        v16bf a1 = *reinterpret_cast<const v16bf*>(A1 + kc * 32);
        v16bf b0 = *reinterpret_cast<const v16bf*>(B0 + kc * 32);
        v16bf b1 = *reinterpret_cast<const v16bf*>(B1 + kc * 32);
        acc00 = wmma_bf16(a0, b0, acc00);
        acc01 = wmma_bf16(a0, b1, acc01);
        acc10 = wmma_bf16(a1, b0, acc10);
        acc11 = wmma_bf16(a1, b1, acc11);
    }
    const int   n   = m;
    const float bv0 = bias[c0 + n];
    const float bv1 = bias[c0 + 16 + n];
#pragma unroll
    for (int r = 0; r < 8; ++r) {
        const int M = r + 8 * half;
        Hbf[(row0 + M)      * DVQ + c0 + n]      = f2bf(acc00[r] + bv0);
        Hbf[(row0 + M)      * DVQ + c0 + 16 + n] = f2bf(acc01[r] + bv1);
        Hbf[(row0 + 16 + M) * DVQ + c0 + n]      = f2bf(acc10[r] + bv0);
        Hbf[(row0 + 16 + M) * DVQ + c0 + 16 + n] = f2bf(acc11[r] + bv1);
    }
}

// ------- K4: score GEMM + fused argmax; async-staged A (64 rows), 4 row tiles --
__global__ void __launch_bounds__(256)
k_score_argmax(const unsigned short* __restrict__ Hbf, const unsigned short* __restrict__ Cbf,
               const float* __restrict__ cnorm, int* __restrict__ idxbuf) {
    extern __shared__ unsigned short dynsmem[];
    const int tid  = threadIdx.x;
    const int wave = tid >> 5, lane = tid & 31;
    const int m    = lane & 15, half = lane >> 4;
    const size_t row0 = (size_t)blockIdx.x * 64;

    { // async stage A panel: 64 rows x 2048 B -> rows padded to 2080 B
        const unsigned lbase = (unsigned)(uintptr_t)dynsmem;
#pragma unroll 4
        for (int i = 0; i < 32; ++i) {
            const int c   = i * 256 + tid;          // 8192 x 16B chunks
            const int r   = c >> 7;
            const int off = (c & 127) * 16;
            async_b128(lbase + (unsigned)(r * (ASTRIDE_D * 2) + off),
                       (const char*)(Hbf + (row0 + r) * DVQ) + off);
        }
        async_wait0();
    }
    __syncthreads();

    const unsigned short* Ap[4];
#pragma unroll
    for (int rt = 0; rt < 4; ++rt)
        Ap[rt] = dynsmem + (size_t)(rt * 16 + m) * ASTRIDE_D + half * 16;

    float best[4][8]; int bidx[4][8];
#pragma unroll
    for (int rt = 0; rt < 4; ++rt)
#pragma unroll
        for (int r = 0; r < 8; ++r) { best[rt][r] = -3.4e38f; bidx[rt][r] = 0; }

    for (int kt = wave; kt < KCB / 16; kt += 8) {
        const int k0 = kt * 16;
        const unsigned short* Bp = Cbf + ((size_t)(k0 + m)) * DVQ + half * 16;
        v8f c0t = {0.f,0.f,0.f,0.f,0.f,0.f,0.f,0.f}, c1t = c0t, c2t = c0t, c3t = c0t;
#pragma unroll 2
        for (int dc = 0; dc < DVQ / 32; ++dc) {
            v16bf b  = *reinterpret_cast<const v16bf*>(Bp + dc * 32);
            v16bf a0 = *reinterpret_cast<const v16bf*>(Ap[0] + dc * 32);
            v16bf a1 = *reinterpret_cast<const v16bf*>(Ap[1] + dc * 32);
            v16bf a2 = *reinterpret_cast<const v16bf*>(Ap[2] + dc * 32);
            v16bf a3 = *reinterpret_cast<const v16bf*>(Ap[3] + dc * 32);
            c0t = wmma_bf16(a0, b, c0t);
            c1t = wmma_bf16(a1, b, c1t);
            c2t = wmma_bf16(a2, b, c2t);
            c3t = wmma_bf16(a3, b, c3t);
        }
        const int   kcol = k0 + m;
        const float cn   = cnorm[kcol];
        v8f* accs[4] = {&c0t, &c1t, &c2t, &c3t};
#pragma unroll
        for (int rt = 0; rt < 4; ++rt)
#pragma unroll
            for (int r = 0; r < 8; ++r) {
                const float v = (*accs[rt])[r] - cn;
                if (v > best[rt][r] || (v == best[rt][r] && kcol < bidx[rt][r])) {
                    best[rt][r] = v; bidx[rt][r] = kcol;
                }
            }
    }
    // reduce across the 16 columns (lanes within each half)
#pragma unroll
    for (int off = 8; off > 0; off >>= 1) {
#pragma unroll
        for (int rt = 0; rt < 4; ++rt)
#pragma unroll
            for (int r = 0; r < 8; ++r) {
                const float ov = __shfl_xor(best[rt][r], off, 32);
                const int   oi = __shfl_xor(bidx[rt][r], off, 32);
                if (ov > best[rt][r] || (ov == best[rt][r] && oi < bidx[rt][r])) {
                    best[rt][r] = ov; bidx[rt][r] = oi;
                }
            }
    }
    __syncthreads();   // done reading A panel; reuse LDS for the reduction
    float* svl = reinterpret_cast<float*>(dynsmem);          // [8][64]
    int*   sid = reinterpret_cast<int*>((char*)dynsmem + 4096);
    if (m == 0) {
#pragma unroll
        for (int rt = 0; rt < 4; ++rt)
#pragma unroll
            for (int r = 0; r < 8; ++r) {
                const int M = rt * 16 + r + 8 * half;
                svl[wave * 64 + M] = best[rt][r];
                sid[wave * 64 + M] = bidx[rt][r];
            }
    }
    __syncthreads();
    if (tid < 64) {
        float bv = svl[tid]; int bk = sid[tid];
#pragma unroll
        for (int w = 1; w < 8; ++w) {
            const float ov = svl[w * 64 + tid]; const int oi = sid[w * 64 + tid];
            if (ov > bv || (ov == bv && oi < bk)) { bv = ov; bk = oi; }
        }
        idxbuf[blockIdx.x * 64 + tid] = bk;
    }
}

// ------- K5: gather quant rows (bf16) + per-row loss partial + indices out ------
__global__ void k_gather_loss(const int* __restrict__ idxbuf, const float* __restrict__ cb,
                              const unsigned short* __restrict__ Hbf,
                              unsigned short* __restrict__ Qbf, float* __restrict__ lossp,
                              float* __restrict__ out_idx) {
    __shared__ float red[256];
    const int bt = blockIdx.x;
    const int k  = idxbuf[bt];
    const float*          crow = cb  + (size_t)k  * DVQ;
    const unsigned short* hrow = Hbf + (size_t)bt * DVQ;
    unsigned short*       qrow = Qbf + (size_t)bt * DVQ;
    float s = 0.f;
    for (int i = threadIdx.x; i < DVQ; i += 256) {
        const float cv = crow[i];
        qrow[i] = f2bf(cv);
        const float d = bf2f(hrow[i]) - cv;
        s += d * d;
    }
    red[threadIdx.x] = s;
    __syncthreads();
    for (int off = 128; off > 0; off >>= 1) {
        if (threadIdx.x < off) red[threadIdx.x] += red[threadIdx.x + off];
        __syncthreads();
    }
    if (threadIdx.x == 0) {
        lossp[bt]   = red[0];
        out_idx[bt] = (float)k;
    }
}

// ---------------- K6: deterministic loss reduction ----------------
__global__ void k_loss_reduce(const float* __restrict__ lossp, float* __restrict__ out_loss) {
    __shared__ float red[256];
    float s = 0.f;
    for (int i = threadIdx.x; i < B_ * T2; i += 256) s += lossp[i];
    red[threadIdx.x] = s;
    __syncthreads();
    for (int off = 128; off > 0; off >>= 1) {
        if (threadIdx.x < off) red[threadIdx.x] += red[threadIdx.x + off];
        __syncthreads();
    }
    if (threadIdx.x == 0)
        out_loss[0] = red[0] / ((float)B_ * (float)T2 * (float)DVQ);
}

// -------- K7: out GEMM (transposed: rows=t2, cols=o); async-staged Q panel -----
// block: 256 thr (8 waves). Covers 32 t2-rows x 256 o-cols for one b.
__global__ void __launch_bounds__(256)
k_gemm_out(const unsigned short* __restrict__ Obf, const unsigned short* __restrict__ Qbf,
           const float* __restrict__ outb, const float* __restrict__ xmask,
           float* __restrict__ out) {
    extern __shared__ unsigned short dynsmem[];
    const int tid  = threadIdx.x;
    const int wave = tid >> 5, lane = tid & 31;
    const int m    = lane & 15, half = lane >> 4;
    const int b  = blockIdx.z;
    const int t0 = blockIdx.y * 32;                  // t2 base
    const int c0 = blockIdx.x * 256 + wave * 32;     // o base for this wave

    { // async stage Q panel: 32 rows x 2048 B -> rows padded to 2080 B
        const unsigned lbase = (unsigned)(uintptr_t)dynsmem;
#pragma unroll 4
        for (int i = 0; i < 16; ++i) {
            const int c   = i * 256 + tid;           // 4096 x 16B chunks
            const int r   = c >> 7;
            const int off = (c & 127) * 16;
            async_b128(lbase + (unsigned)(r * (ASTRIDE_D * 2) + off),
                       (const char*)(Qbf + ((size_t)(b * T2 + t0 + r)) * DVQ) + off);
        }
        async_wait0();
    }
    __syncthreads();

    const unsigned short* A0 = dynsmem + (size_t)(0  + m) * ASTRIDE_D + half * 16;
    const unsigned short* A1 = dynsmem + (size_t)(16 + m) * ASTRIDE_D + half * 16;
    const unsigned short* B0 = Obf + ((size_t)(c0 + m))      * DVQ + half * 16;
    const unsigned short* B1 = Obf + ((size_t)(c0 + 16 + m)) * DVQ + half * 16;

    v8f acc00 = {0.f,0.f,0.f,0.f,0.f,0.f,0.f,0.f}, acc01 = acc00, acc10 = acc00, acc11 = acc00;
#pragma unroll 2
    for (int dc = 0; dc < DVQ / 32; ++dc) {
        v16bf a0 = *reinterpret_cast<const v16bf*>(A0 + dc * 32);
        v16bf a1 = *reinterpret_cast<const v16bf*>(A1 + dc * 32);
        v16bf b0 = *reinterpret_cast<const v16bf*>(B0 + dc * 32);
        v16bf b1 = *reinterpret_cast<const v16bf*>(B1 + dc * 32);
        acc00 = wmma_bf16(a0, b0, acc00);
        acc01 = wmma_bf16(a0, b1, acc01);
        acc10 = wmma_bf16(a1, b0, acc10);
        acc11 = wmma_bf16(a1, b1, acc11);
    }
    const int   n   = m;
    const float bo0 = outb[c0 + n];
    const float bo1 = outb[c0 + 16 + n];
    const v8f*  accA[2] = {&acc00, &acc10};   // col tile 0, row tiles 0/1
    const v8f*  accB[2] = {&acc01, &acc11};   // col tile 1
#pragma unroll
    for (int rt = 0; rt < 2; ++rt) {
#pragma unroll
        for (int r = 0; r < 8; ++r) {
            const int t2g = t0 + rt * 16 + r + 8 * half;
            const float m0 = xmask[(size_t)b * T_ + 2 * t2g];
            const float m1 = xmask[(size_t)b * T_ + 2 * t2g + 1];
            const float v0 = (*accA[rt])[r] + bo0;
            const float v1 = (*accB[rt])[r] + bo1;
            float2* d0 = reinterpret_cast<float2*>(out + ((size_t)(b * CIN + c0 + n))      * T_ + 2 * t2g);
            float2* d1 = reinterpret_cast<float2*>(out + ((size_t)(b * CIN + c0 + 16 + n)) * T_ + 2 * t2g);
            *d0 = make_float2(v0 * m0, v0 * m1);
            *d1 = make_float2(v1 * m0, v1 * m1);
        }
    }
}

// ---------------- launch ----------------
extern "C" void kernel_launch(void* const* d_in, const int* in_sizes, int n_in,
                              void* d_out, int out_size, void* d_ws, size_t ws_size,
                              hipStream_t stream) {
    (void)in_sizes; (void)n_in; (void)out_size; (void)ws_size;
    const float* x      = (const float*)d_in[0];
    const float* xmask  = (const float*)d_in[1];
    const float* win    = (const float*)d_in[2];
    const float* binb   = (const float*)d_in[3];
    const float* cbook  = (const float*)d_in[4];
    const float* wout   = (const float*)d_in[5];
    const float* outb   = (const float*)d_in[6];
    float* out = (float*)d_out;

    char* ws = (char*)d_ws;
    unsigned short* Xbf   = (unsigned short*)(ws);                         // 64 MB
    unsigned short* Wbf   = (unsigned short*)(ws + 67108864);              //  4 MB
    unsigned short* Cbf   = (unsigned short*)(ws + 71303168);              //  4 MB
    unsigned short* Obf   = (unsigned short*)(ws + 75497472);              //  2 MB
    unsigned short* Hbf   = (unsigned short*)(ws + 77594624);              // 32 MB
    unsigned short* Qbf   = (unsigned short*)(ws + 111149056);             // 32 MB
    float*          cnorm = (float*)(ws + 144703488);                      //  8 KB
    int*            idxb  = (int*)(ws + 144711680);                        // 64 KB
    float*          lossp = (float*)(ws + 144777216);                      // 64 KB

    float* out_idx  = out + (size_t)B_ * CIN * T_;
    float* out_loss = out_idx + (size_t)B_ * T2;

    const size_t smem_k = (size_t)32 * ASTRIDE_K * 2;   // 132096 B
    const size_t smem_s = (size_t)64 * ASTRIDE_D * 2;   // 133120 B
    const size_t smem_o = (size_t)32 * ASTRIDE_D * 2;   //  66560 B

    k_pack_x      <<<dim3(T2 / 32, CIN / 32, B_), 256, 0, stream>>>(x, Xbf);
    k_cvt_w       <<<(5242880 + 255) / 256, 256, 0, stream>>>(win, cbook, wout, Wbf, Cbf, Obf);
    k_cnorm       <<<KCB, 256, 0, stream>>>(cbook, cnorm);
    k_gemm_convin <<<dim3(DVQ / 256, (B_ * T2) / 32), 256, smem_k, stream>>>(Xbf, Wbf, binb, Hbf);
    k_score_argmax<<<(B_ * T2) / 64, 256, smem_s, stream>>>(Hbf, Cbf, cnorm, idxb);
    k_gather_loss <<<B_ * T2, 256, 0, stream>>>(idxb, cbook, Hbf, Qbf, lossp, out_idx);
    k_loss_reduce <<<1, 256, 0, stream>>>(lossp, out_loss);
    k_gemm_out    <<<dim3(CIN / 256, T2 / 32, B_), 256, smem_o, stream>>>(Obf, Qbf, outb, xmask, out);
}
